// Histoformer_63909113364891
// MI455X (gfx1250) — compile-verified, matrix operand order checked by reading, not run on hardware
//
#include <hip/hip_runtime.h>
#include <hip/hip_bf16.h>

// ---------------- problem constants ----------------
constexpr int  BB     = 2;
constexpr int  CDIM   = 192;
constexpr int  IH     = 128;
constexpr int  IW     = 128;
constexpr int  NHW    = IH * IW;            // 16384
constexpr long CHWL   = (long)CDIM * NHW;   // 3145728
constexpr int  NHEADS = 4;
constexpr int  CPD    = CDIM / NHEADS;      // 48
constexpr int  HWQ    = NHW / NHEADS;       // 4096
constexpr int  NHID   = 512;
constexpr int  HALF   = CDIM / 2;           // 96

typedef __attribute__((ext_vector_type(16))) _Float16 v16h;
typedef __attribute__((ext_vector_type(8)))  _Float16 v8h;
typedef __attribute__((ext_vector_type(4)))  _Float16 v4h;
typedef __attribute__((ext_vector_type(8)))  float    v8f;

// ======================================================================
// Double-buffered LDS-tiled WMMA GEMM: C[M,N] = A[M,K]*op(B) (+R).
//   BT=false: B is [K,N] row-major (NN).  BT=true: B is [N,K] row-major (NT).
// REQUIRES: M%64==0, N%64==0, K%32==0 (true for every call site).
// Block = 128 threads (4 waves) -> 64x64 C tile; wave w owns rows w*16..+15.
// Software pipeline: global loads for tile t+1 issued before the WMMA chain
// of tile t; convert+LDS-store after compute; ONE barrier per K-step.
// ======================================================================
template <bool BT>
__global__ void __launch_bounds__(128)
gemm_wmma(const float* __restrict__ A, const float* __restrict__ B,
          float* __restrict__ C, const float* __restrict__ R,
          int M, int N, int K, int lda, int ldb, int ldc,
          long sA, long sB, long sC, long sR)
{
    __shared__ __align__(16) _Float16 As[2][64][32];
    __shared__ __align__(16) _Float16 Bs[2][64][32];   // [n][k] (K-contiguous)

    const int tid    = threadIdx.x;
    const int lane   = tid & 31;
    const int wv     = tid >> 5;
    const int n_base = blockIdx.x * 64;
    const int m_base = blockIdx.y * 64;
    const int bz     = blockIdx.z;

    const float* Ab = A + bz * sA;
    const float* Bb = B + bz * sB;

    // staging roles (unconditional, coalesced float4 loads)
    const int ar = tid >> 1;            // A row 0..63 (also B n-row for NT)
    const int ak = (tid & 1) * 16;      // k-offset 0/16
    const int bk = tid >> 2;            // B k-row 0..31   (NN)
    const int bn = (tid & 3) * 16;      // B col-offset    (NN)

    // fragment roles (ISA 7.12.2 16-bit layout)
    const int fr = lane & 15;           // row (A) / col (B) within 16-tile
    const int kh = (lane >> 4) * 8;     // K-half select

    float4 ra[4], rb[4];

    auto load_tile = [&](int k0) {
        const float* ap = Ab + (long)(m_base + ar) * lda + k0 + ak;
        if (k0 + 64 < K) __builtin_prefetch(ap + 64, 0, 0);   // two tiles ahead
#pragma unroll
        for (int i = 0; i < 4; ++i) ra[i] = *(const float4*)(ap + 4 * i);
        if (BT) {
            const float* bp = Bb + (long)(n_base + ar) * ldb + k0 + ak;
            if (k0 + 64 < K) __builtin_prefetch(bp + 64, 0, 0);
#pragma unroll
            for (int i = 0; i < 4; ++i) rb[i] = *(const float4*)(bp + 4 * i);
        } else {
            const float* bp = Bb + (long)(k0 + bk) * ldb + n_base + bn;
            if (k0 + 64 < K) __builtin_prefetch(bp + (long)64 * ldb, 0, 0);
#pragma unroll
            for (int i = 0; i < 4; ++i) rb[i] = *(const float4*)(bp + 4 * i);
        }
    };

    auto store_tile = [&](int buf) {
#pragma unroll
        for (int i = 0; i < 4; ++i) {
            v4h hq = { (_Float16)ra[i].x, (_Float16)ra[i].y,
                       (_Float16)ra[i].z, (_Float16)ra[i].w };
            *(v4h*)&As[buf][ar][ak + 4 * i] = hq;
        }
        if (BT) {
#pragma unroll
            for (int i = 0; i < 4; ++i) {
                v4h hq = { (_Float16)rb[i].x, (_Float16)rb[i].y,
                           (_Float16)rb[i].z, (_Float16)rb[i].w };
                *(v4h*)&Bs[buf][ar][ak + 4 * i] = hq;
            }
        } else {
#pragma unroll
            for (int i = 0; i < 4; ++i) {
                Bs[buf][bn + 4 * i + 0][bk] = (_Float16)rb[i].x;
                Bs[buf][bn + 4 * i + 1][bk] = (_Float16)rb[i].y;
                Bs[buf][bn + 4 * i + 2][bk] = (_Float16)rb[i].z;
                Bs[buf][bn + 4 * i + 3][bk] = (_Float16)rb[i].w;
            }
        }
    };

    v8f acc[4] = {};

    // prologue: stage tile 0 into buffer 0
    load_tile(0);
    store_tile(0);
    __syncthreads();

    const int KT = K / 32;
    for (int kt = 0; kt < KT; ++kt) {
        const int  cur  = kt & 1;
        const bool more = (kt + 1 < KT);
        if (more) load_tile((kt + 1) * 32);   // global loads in flight over compute

        // ---- compute from buf[cur]: preload fragments, then 4 WMMAs ----
        const v8h alo = *(const v8h*)&As[cur][wv * 16 + fr][kh];
        const v8h ahi = *(const v8h*)&As[cur][wv * 16 + fr][kh + 16];
        const v16h af = __builtin_shufflevector(alo, ahi,
            0, 1, 2, 3, 4, 5, 6, 7, 8, 9, 10, 11, 12, 13, 14, 15);
        v16h bfr[4];
#pragma unroll
        for (int j = 0; j < 4; ++j) {
            const v8h blo = *(const v8h*)&Bs[cur][j * 16 + fr][kh];
            const v8h bhi = *(const v8h*)&Bs[cur][j * 16 + fr][kh + 16];
            bfr[j] = __builtin_shufflevector(blo, bhi,
                0, 1, 2, 3, 4, 5, 6, 7, 8, 9, 10, 11, 12, 13, 14, 15);
        }
#pragma unroll
        for (int j = 0; j < 4; ++j)
            acc[j] = __builtin_amdgcn_wmma_f32_16x16x32_f16(false, af, false, bfr[j],
                                                            (short)0, acc[j], false, false);

        if (more) store_tile(cur ^ 1);        // convert + LDS store after compute
        __syncthreads();                      // single barrier per K-step
    }

    // ---- store (C/D layout: lanes 0-15 rows +0..7, lanes 16-31 rows +8..15) ----
    const int rbase = m_base + wv * 16 + ((lane >> 4) << 3);
#pragma unroll
    for (int j = 0; j < 4; ++j) {
        const int col = n_base + j * 16 + (lane & 15);
#pragma unroll
        for (int j2 = 0; j2 < 8; ++j2) {
            const long o = bz * sC + (long)(rbase + j2) * ldc + col;
            float v = acc[j][j2];
            if (R) v += R[bz * sR + (long)(rbase + j2) * ldc + col];
            C[o] = v;
        }
    }
}

// ======================================================================
// LayerNorm over channel dim (per pixel).
// ======================================================================
__global__ void ln_kernel(const float* __restrict__ x, float* __restrict__ y,
                          int C, int HWn)
{
    const int b = blockIdx.y;
    const int p = blockIdx.x * blockDim.x + threadIdx.x;
    if (p >= HWn) return;
    const float* xp = x + (long)b * C * HWn + p;
    float mu = 0.0f;
    for (int c = 0; c < C; ++c) mu += xp[(long)c * HWn];
    mu /= (float)C;
    float var = 0.0f;
    for (int c = 0; c < C; ++c) { float d = xp[(long)c * HWn] - mu; var += d * d; }
    var /= (float)C;
    const float inv = rsqrtf(var + 1e-5f);
    float* yp = y + (long)b * C * HWn + p;
    for (int c = 0; c < C; ++c) yp[(long)c * HWn] = (xp[(long)c * HWn] - mu) * inv;
}

// ======================================================================
// In-place bitonic sort of 128-element lines + argsort indices (stable order).
// ======================================================================
__global__ void __launch_bounds__(128)
sort128_kernel(float* __restrict__ data, int* __restrict__ idx_out,
               int n1, int n2, long s0, long s1, long s2, int S,
               long i0, long i1, long i2, int iS)
{
    __shared__ float key[128];
    __shared__ int   kid[128];
    const int line  = blockIdx.x;
    const int outer = line / (n1 * n2);
    const int mid   = (line / n2) % n1;
    const int inner = line % n2;
    const long base  = (long)outer * s0 + (long)mid * s1 + (long)inner * s2;
    const long ibase = (long)outer * i0 + (long)mid * i1 + (long)inner * i2;
    const int t = threadIdx.x;
    key[t] = data[base + (long)t * S];
    kid[t] = t;
    __syncthreads();
    for (int k = 2; k <= 128; k <<= 1) {
        for (int j = k >> 1; j > 0; j >>= 1) {
            const int p = t ^ j;
            if (p > t) {
                const bool up = ((t & k) == 0);
                float a = key[t], b = key[p];
                int   ia = kid[t], ib = kid[p];
                const bool gt = (a > b) || (a == b && ia > ib);
                if (gt == up) { key[t] = b; key[p] = a; kid[t] = ib; kid[p] = ia; }
            }
            __syncthreads();
        }
    }
    data[base + (long)t * S]      = key[t];
    idx_out[ibase + (long)t * iS] = kid[t];
}

// ======================================================================
// Bitonic sort of 16384-element lines in 128KB dynamic LDS (320KB/WGP).
// ======================================================================
__global__ void sort16k_kernel(const float* __restrict__ in,
                               float* __restrict__ vals, int* __restrict__ idx_out,
                               int n1, long s0, long s1)
{
    extern __shared__ char smem[];
    constexpr int n = NHW;
    float* key = (float*)smem;
    int*   kid = (int*)(smem + sizeof(float) * n);
    const int  line = blockIdx.x;
    const long base = (long)(line / n1) * s0 + (long)(line % n1) * s1;
    const int tid = threadIdx.x, nt = blockDim.x;
    for (int t = tid; t < n; t += nt) { key[t] = in[base + t]; kid[t] = t; }
    __syncthreads();
    for (int k = 2; k <= n; k <<= 1) {
        for (int j = k >> 1; j > 0; j >>= 1) {
            for (int t = tid; t < n; t += nt) {
                const int p = t ^ j;
                if (p > t) {
                    const bool up = ((t & k) == 0);
                    float a = key[t], b = key[p];
                    int   ia = kid[t], ib = kid[p];
                    const bool gt = (a > b) || (a == b && ia > ib);
                    if (gt == up) { key[t] = b; key[p] = a; kid[t] = ib; kid[p] = ia; }
                }
            }
            __syncthreads();
        }
    }
    for (int t = tid; t < n; t += nt) {
        vals[(long)line * n + t]    = key[t];
        idx_out[(long)line * n + t] = kid[t];
    }
}

// ======================================================================
// Depthwise conv (square kernel, dilation), explicit batch strides.
// ======================================================================
__global__ void dwconv_kernel(const float* __restrict__ in, const float* __restrict__ w,
                              float* __restrict__ out, int C, int H, int W,
                              int ks, int pad, int dil, long sIn, long sOut)
{
    const int p = blockIdx.x * blockDim.x + threadIdx.x;
    if (p >= H * W) return;
    const int c = blockIdx.y, b = blockIdx.z;
    const int h = p / W, x0 = p % W;
    const float* ip = in + (long)b * sIn + (long)c * H * W;
    const float* wp = w + (long)c * ks * ks;
    float acc = 0.0f;
    for (int i = 0; i < ks; ++i) {
        const int hh = h + i * dil - pad;
        if (hh < 0 || hh >= H) continue;
        for (int j = 0; j < ks; ++j) {
            const int ww = x0 + j * dil - pad;
            if (ww < 0 || ww >= W) continue;
            acc += ip[(long)hh * W + ww] * wp[i * ks + j];
        }
    }
    out[(long)b * sOut + (long)c * H * W + p] = acc;
}

// ======================================================================
// Gather (by idx along L) + fwd head-reshape -> (B,HEADS,192,4096).
// ======================================================================
__global__ void build_box_kernel(const float* __restrict__ src, const int* __restrict__ idx,
                                 float* __restrict__ dst, long srcBstride, int box)
{
    long t = (long)blockIdx.x * blockDim.x + threadIdx.x;
    const long total = (long)BB * NHEADS * CDIM * HWQ;
    if (t >= total) return;
    const int nn = (int)(t % HWQ);
    long r = t / HWQ;
    const int row = (int)(r % CDIM); r /= CDIM;
    const int h = (int)(r % NHEADS);
    const int b = (int)(r / NHEADS);
    const int cp = row >> 2, f = row & 3;
    const int c  = h * CPD + cp;
    const int j  = box ? (f * HWQ + nn) : (nn * NHEADS + f);
    const int pos = idx ? idx[((long)b * CDIM + c) * NHW + j] : j;
    dst[t] = src[(long)b * srcBstride + (long)c * NHW + pos];
}

// ======================================================================
// In-place L2 normalization of rows, eps floor 1e-12.
// ======================================================================
__global__ void __launch_bounds__(256)
rownorm_kernel(float* __restrict__ m, int ncols)
{
    __shared__ float red[256];
    const long row = blockIdx.x;
    float* p = m + row * (long)ncols;
    float s = 0.0f;
    for (int i = threadIdx.x; i < ncols; i += blockDim.x) { float v = p[i]; s += v * v; }
    red[threadIdx.x] = s;
    __syncthreads();
    for (int o = 128; o > 0; o >>= 1) {
        if ((int)threadIdx.x < o) red[threadIdx.x] += red[threadIdx.x + o];
        __syncthreads();
    }
    const float inv = 1.0f / fmaxf(sqrtf(red[0]), 1e-12f);
    for (int i = threadIdx.x; i < ncols; i += blockDim.x) p[i] *= inv;
}

// ======================================================================
// softmax_1 with per-head temperature:  e = exp(a*temp); e / (sum(e)+1)
// ======================================================================
__global__ void __launch_bounds__(256)
softmax1_kernel(float* __restrict__ a, const float* __restrict__ temp, int ncols)
{
    __shared__ float red[256];
    const long row = blockIdx.x;
    const int  h   = (int)((row / CDIM) % NHEADS);
    const float tp = temp[h];
    float* p = a + row * (long)ncols;
    float s = 0.0f;
    for (int i = threadIdx.x; i < ncols; i += blockDim.x) {
        const float e = __expf(p[i] * tp);
        p[i] = e;
        s += e;
    }
    red[threadIdx.x] = s;
    __syncthreads();
    for (int o = 128; o > 0; o >>= 1) {
        if ((int)threadIdx.x < o) red[threadIdx.x] += red[threadIdx.x + o];
        __syncthreads();
    }
    const float denom = 1.0f / (red[0] + 1.0f);
    for (int i = threadIdx.x; i < ncols; i += blockDim.x) p[i] *= denom;
}

// ======================================================================
// inverse head-reshape of both attention outputs, multiply, scatter by idx.
// ======================================================================
__global__ void inv_scatter_mul_kernel(const float* __restrict__ O1, const float* __restrict__ O2,
                                       const int* __restrict__ idx, float* __restrict__ prod)
{
    long t = (long)blockIdx.x * blockDim.x + threadIdx.x;
    const long total = (long)BB * CHWL;
    if (t >= total) return;
    const int j = (int)(t % NHW);
    long r = t / NHW;
    const int c = (int)(r % CDIM);
    const int b = (int)(r / CDIM);
    const int h = c / CPD, cp = c % CPD;
    const long bh = (long)(b * NHEADS + h) * CDIM;
    const int f1 = j / HWQ, n1 = j % HWQ;                 // box inverse
    const float v1 = O1[(bh + (cp * 4 + f1)) * HWQ + n1];
    const int f2 = j % NHEADS, n2 = j / NHEADS;           // non-box inverse
    const float v2 = O2[(bh + (cp * 4 + f2)) * HWQ + n2];
    const int pos = idx[((long)b * CDIM + c) * NHW + j];
    prod[((long)b * CDIM + c) * NHW + pos] = v1 * v2;
}

// un-sort along W (first HALF channels)
__global__ void scatter_w_kernel(const float* __restrict__ src, const int* __restrict__ idx_w,
                                 float* __restrict__ dst)
{
    long t = (long)blockIdx.x * blockDim.x + threadIdx.x;
    const long total = (long)BB * HALF * NHW;
    if (t >= total) return;
    const int w = (int)(t % IW);
    long r = t / IW;
    const int h = (int)(r % IH); r /= IH;
    const int c = (int)(r % HALF);
    const int b = (int)(r / HALF);
    const long hl = ((long)(b * HALF + c) * IH + h) * IW;
    const int dw = idx_w[hl + w];
    dst[hl + dw] = src[((long)b * CDIM + c) * NHW + (long)h * IW + w];
}

// un-sort along H (first HALF) + residual; second half: plain residual add
__global__ void unsort_res_kernel(const float* __restrict__ tmpW, const int* __restrict__ idx_h,
                                  const float* __restrict__ x, const float* __restrict__ projo,
                                  float* __restrict__ y)
{
    long t = (long)blockIdx.x * blockDim.x + threadIdx.x;
    const long total = (long)BB * CHWL;
    if (t >= total) return;
    const int w = (int)(t % IW);
    long r = t / IW;
    const int h = (int)(r % IH); r /= IH;
    const int c = (int)(r % CDIM);
    const int b = (int)(r / CDIM);
    if (c < HALF) {
        const long hl = ((long)(b * HALF + c) * IH + h) * IW + w;
        const int dh = idx_h[hl];
        const long od = ((long)b * CDIM + c) * NHW + (long)dh * IW + w;
        y[od] = x[od] + tmpW[((long)(b * HALF + c) * IH + h) * IW + w];
    } else {
        const long o = ((long)b * CDIM + c) * NHW + (long)h * IW + w;
        y[o] = x[o] + projo[o];
    }
}

// pixel_shuffle r=2: (B,1024,128,128) -> (B,256,256,256)
__global__ void pixel_shuffle_kernel(const float* __restrict__ in, float* __restrict__ out)
{
    long t = (long)blockIdx.x * blockDim.x + threadIdx.x;
    const long total = (long)BB * 1024 * NHW;
    if (t >= total) return;
    const int wo = (int)(t % 256);
    long r = t / 256;
    const int ho = (int)(r % 256); r /= 256;
    const int co = (int)(r % 256);
    const int b  = (int)(r / 256);
    const int h = ho >> 1, r1 = ho & 1, w = wo >> 1, r2 = wo & 1;
    out[t] = in[(((long)b * 1024 + (co * 4 + r1 * 2 + r2)) * IH + h) * IW + w];
}

// mish-gate + pixel_unshuffle r=2 -> (B,512,128,128)
__global__ void gate_unshuffle_kernel(const float* __restrict__ d1, const float* __restrict__ d2,
                                      float* __restrict__ u)
{
    long t = (long)blockIdx.x * blockDim.x + threadIdx.x;
    const long total = (long)BB * 512 * NHW;
    if (t >= total) return;
    const int w = (int)(t % IW);
    long r = t / IW;
    const int h  = (int)(r % IH); r /= IH;
    const int cu = (int)(r % 512);
    const int b  = (int)(r / 512);
    const int c = cu >> 2, r1 = (cu >> 1) & 1, r2 = cu & 1;
    const long s = (((long)b * 128 + c) * 256 + (h * 2 + r1)) * 256 + (w * 2 + r2);
    const float x1 = d1[s], x2 = d2[s];
    const float sp = (x2 > 20.0f) ? x2 : log1pf(__expf(x2));
    u[t] = x2 * tanhf(sp) * x1;
}

// ======================================================================
// host-side orchestration
// ======================================================================
extern "C" void kernel_launch(void* const* d_in, const int* in_sizes, int n_in,
                              void* d_out, int out_size, void* d_ws, size_t ws_size,
                              hipStream_t stream)
{
    (void)in_sizes; (void)n_in; (void)out_size; (void)ws_size;
    const float* x        = (const float*)d_in[0];
    const float* temp     = (const float*)d_in[1];
    const float* w_qkv    = (const float*)d_in[2];
    const float* w_qkv_dw = (const float*)d_in[3];
    const float* w_proj   = (const float*)d_in[4];
    const float* w_ffn_in = (const float*)d_in[5];
    const float* w_dw5    = (const float*)d_in[6];
    const float* w_dwd    = (const float*)d_in[7];
    const float* w_ffn_out= (const float*)d_in[8];
    float* out = (float*)d_out;

    // bump allocator over workspace (~1.14 GB total)
    char* ws = (char*)d_ws;
    size_t off = 0;
    auto allocf = [&](long n) -> float* {
        float* p = (float*)(ws + off);
        off = (off + (size_t)n * sizeof(float) + 255) & ~(size_t)255;
        return p;
    };
    auto alloci = [&](long n) -> int* {
        int* p = (int*)(ws + off);
        off = (off + (size_t)n * sizeof(int) + 255) & ~(size_t)255;
        return p;
    };

    float* t_ln  = allocf((long)BB * CHWL);
    int*   idx_h = alloci((long)BB * HALF * NHW);
    int*   idx_w = alloci((long)BB * HALF * NHW);
    float* qkv   = allocf((long)BB * 5 * CHWL);
    float* qkvd  = allocf((long)BB * 5 * CHWL);
    float* vs    = allocf((long)BB * CHWL);
    int*   idx   = alloci((long)BB * CHWL);
    float* Qb = allocf((long)BB * CHWL);
    float* Kb = allocf((long)BB * CHWL);
    float* Vb = allocf((long)BB * CHWL);
    float* Qn = allocf((long)BB * CHWL);
    float* Kn = allocf((long)BB * CHWL);
    float* Vn = allocf((long)BB * CHWL);
    float* attn1 = allocf((long)BB * NHEADS * CDIM * CDIM);
    float* attn2 = allocf((long)BB * NHEADS * CDIM * CDIM);
    float* O1    = allocf((long)BB * CHWL);
    float* O2    = allocf((long)BB * CHWL);
    float* prod  = allocf((long)BB * CHWL);
    float* projo = allocf((long)BB * CHWL);
    float* tmpW  = allocf((long)BB * HALF * NHW);
    float* y1    = allocf((long)BB * CHWL);
    float* ln2b  = allocf((long)BB * CHWL);
    float* big   = allocf((long)BB * 1024 * NHW);
    float* shuf  = allocf((long)BB * 1024 * NHW);
    float* dd1   = allocf((long)BB * 128 * 65536);
    float* dd2   = allocf((long)BB * 128 * 65536);
    float* ub    = allocf((long)BB * 512 * NHW);

    auto blocks = [](long total) { return dim3((unsigned)((total + 255) / 256)); };

    // ---- LN1 ----
    ln_kernel<<<dim3((NHW + 255) / 256, BB), 256, 0, stream>>>(x, t_ln, CDIM, NHW);

    // ---- sort first HALF channels along H then W (in place), recording indices ----
    sort128_kernel<<<BB * HALF * IW, 128, 0, stream>>>(
        t_ln, idx_h, HALF, IW, (long)CDIM * NHW, (long)NHW, 1L, IW,
        (long)HALF * NHW, (long)NHW, 1L, IW);
    sort128_kernel<<<BB * HALF * IH, 128, 0, stream>>>(
        t_ln, idx_w, HALF, IH, (long)CDIM * NHW, (long)NHW, (long)IW, 1,
        (long)HALF * NHW, (long)NHW, (long)IW, 1);

    // ---- qkv 1x1 conv: [960,192] x [192,16384] ----
    gemm_wmma<false><<<dim3(NHW / 64, 960 / 64, BB), 128, 0, stream>>>(
        w_qkv, t_ln, qkv, nullptr, 960, NHW, CDIM, CDIM, NHW, NHW,
        0L, CHWL, 5L * CHWL, 0L);

    // ---- depthwise 3x3 over 960 channels ----
    dwconv_kernel<<<dim3((NHW + 255) / 256, 960, BB), 256, 0, stream>>>(
        qkv, w_qkv_dw, qkvd, 960, IH, IW, 3, 1, 1, 5L * CHWL, 5L * CHWL);

    // ---- argsort v along L=16384 (bitonic in 128KB LDS) ----
    sort16k_kernel<<<BB * CDIM, 256, NHW * 8, stream>>>(
        qkvd + 4L * CHWL, vs, idx, CDIM, 5L * CHWL, (long)NHW);

    // ---- gather + head reshapes ----
    const long tot_box = (long)BB * CHWL;
    build_box_kernel<<<blocks(tot_box), 256, 0, stream>>>(qkvd + 0L * CHWL, idx, Qb, 5L * CHWL, 1);
    build_box_kernel<<<blocks(tot_box), 256, 0, stream>>>(qkvd + 1L * CHWL, idx, Kb, 5L * CHWL, 1);
    build_box_kernel<<<blocks(tot_box), 256, 0, stream>>>(vs,               nullptr, Vb, CHWL, 1);
    build_box_kernel<<<blocks(tot_box), 256, 0, stream>>>(qkvd + 2L * CHWL, idx, Qn, 5L * CHWL, 0);
    build_box_kernel<<<blocks(tot_box), 256, 0, stream>>>(qkvd + 3L * CHWL, idx, Kn, 5L * CHWL, 0);
    build_box_kernel<<<blocks(tot_box), 256, 0, stream>>>(vs,               nullptr, Vn, CHWL, 0);

    // ---- L2 normalize rows of Q, K ----
    rownorm_kernel<<<BB * NHEADS * CDIM, 256, 0, stream>>>(Qb, HWQ);
    rownorm_kernel<<<BB * NHEADS * CDIM, 256, 0, stream>>>(Kb, HWQ);
    rownorm_kernel<<<BB * NHEADS * CDIM, 256, 0, stream>>>(Qn, HWQ);
    rownorm_kernel<<<BB * NHEADS * CDIM, 256, 0, stream>>>(Kn, HWQ);

    // ---- attention: logits (NT), softmax_1, apply (NN); batch = B*HEADS ----
    const long sQ = (long)CDIM * HWQ, sAt = (long)CDIM * CDIM;
    gemm_wmma<true><<<dim3(CDIM / 64, CDIM / 64, BB * NHEADS), 128, 0, stream>>>(
        Qb, Kb, attn1, nullptr, CDIM, CDIM, HWQ, HWQ, HWQ, CDIM, sQ, sQ, sAt, 0L);
    gemm_wmma<true><<<dim3(CDIM / 64, CDIM / 64, BB * NHEADS), 128, 0, stream>>>(
        Qn, Kn, attn2, nullptr, CDIM, CDIM, HWQ, HWQ, HWQ, CDIM, sQ, sQ, sAt, 0L);
    softmax1_kernel<<<BB * NHEADS * CDIM, 256, 0, stream>>>(attn1, temp, CDIM);
    softmax1_kernel<<<BB * NHEADS * CDIM, 256, 0, stream>>>(attn2, temp, CDIM);
    gemm_wmma<false><<<dim3(HWQ / 64, CDIM / 64, BB * NHEADS), 128, 0, stream>>>(
        attn1, Vb, O1, nullptr, CDIM, HWQ, CDIM, CDIM, HWQ, HWQ, sAt, sQ, sQ, 0L);
    gemm_wmma<false><<<dim3(HWQ / 64, CDIM / 64, BB * NHEADS), 128, 0, stream>>>(
        attn2, Vn, O2, nullptr, CDIM, HWQ, CDIM, CDIM, HWQ, HWQ, sAt, sQ, sQ, 0L);

    // ---- inverse reshape, multiply, scatter back through idx ----
    inv_scatter_mul_kernel<<<blocks((long)BB * CHWL), 256, 0, stream>>>(O1, O2, idx, prod);

    // ---- proj 1x1 ----
    gemm_wmma<false><<<dim3(NHW / 64, CDIM / 64, BB), 128, 0, stream>>>(
        w_proj, prod, projo, nullptr, CDIM, NHW, CDIM, CDIM, NHW, NHW, 0L, CHWL, CHWL, 0L);

    // ---- undo the H/W sorts on first half + residual add ----
    scatter_w_kernel<<<blocks((long)BB * HALF * NHW), 256, 0, stream>>>(projo, idx_w, tmpW);
    unsort_res_kernel<<<blocks((long)BB * CHWL), 256, 0, stream>>>(tmpW, idx_h, x, projo, y1);

    // ---- FFN ----
    ln_kernel<<<dim3((NHW + 255) / 256, BB), 256, 0, stream>>>(y1, ln2b, CDIM, NHW);
    gemm_wmma<false><<<dim3(NHW / 64, 1024 / 64, BB), 128, 0, stream>>>(
        w_ffn_in, ln2b, big, nullptr, 1024, NHW, CDIM, CDIM, NHW, NHW,
        0L, CHWL, 1024L * NHW, 0L);
    pixel_shuffle_kernel<<<blocks((long)BB * 1024 * NHW), 256, 0, stream>>>(big, shuf);
    dwconv_kernel<<<dim3((256 * 256 + 255) / 256, 128, BB), 256, 0, stream>>>(
        shuf, w_dw5, dd1, 128, 256, 256, 5, 2, 1, 1024L * NHW, 128L * 65536);
    dwconv_kernel<<<dim3((256 * 256 + 255) / 256, 128, BB), 256, 0, stream>>>(
        shuf + 128L * 65536, w_dwd, dd2, 128, 256, 256, 3, 2, 2, 1024L * NHW, 128L * 65536);
    gate_unshuffle_kernel<<<blocks((long)BB * 512 * NHW), 256, 0, stream>>>(dd1, dd2, ub);
    gemm_wmma<false><<<dim3(NHW / 64, CDIM / 64, BB), 128, 0, stream>>>(
        w_ffn_out, ub, out, y1, CDIM, NHW, NHID, NHID, NHW, NHW,
        0L, 512L * NHW, CHWL, CHWL);
}
